// GraphEncoder_49598282334498
// MI455X (gfx1250) — compile-verified, matrix operand order checked by reading, not run on hardware
//
#include <hip/hip_runtime.h>
#include <hip/hip_bf16.h>
#include <math.h>

#define N_NODES 50000
#define N_EDGES 500000
#define N_GRAPHS 64
#define IN_DIM 6
#define FFM_DIM 128
#define GNN_DIM 1024
#define OUT_DIM 256
#define CAT_DIM 2048            // GNN_DIM + 4*2*FFM_DIM
#define M_PAD 50048             // N_NODES rounded up to multiple of 64
#define LN_EPS 1e-5f

// ---------------------------------------------------------------------------
// types for WMMA
// ---------------------------------------------------------------------------
typedef __bf16 bf16x16 __attribute__((ext_vector_type(16)));
typedef float  f32x8   __attribute__((ext_vector_type(8)));

struct U4 { unsigned int x, y, z, w; } __attribute__((aligned(16)));

union FragCvt { U4 q[2]; bf16x16 v; };

__device__ __forceinline__ bf16x16 load_frag(const unsigned short* p0,
                                             const unsigned short* p1) {
    FragCvt f;
    f.q[0] = *reinterpret_cast<const U4*>(p0);
    f.q[1] = *reinterpret_cast<const U4*>(p1);
    return f.v;
}

__device__ __forceinline__ unsigned short f2bf(float f) {
    unsigned int u = __float_as_uint(f);
    unsigned int r = u + 0x7fffu + ((u >> 16) & 1u);   // round-to-nearest-even
    return (unsigned short)(r >> 16);
}
__device__ __forceinline__ float bf2f(unsigned short u) {
    return __uint_as_float(((unsigned int)u) << 16);
}

// ---------------------------------------------------------------------------
// K0: transpose + bf16-convert [W2_l ; W2_r] -> Wt[j][k], j<1024, k<2048
// ---------------------------------------------------------------------------
__global__ void __launch_bounds__(256)
k_prepW(const float* __restrict__ W2l, const float* __restrict__ W2r,
        unsigned short* __restrict__ Wt) {
    size_t idx = (size_t)blockIdx.x * 256 + threadIdx.x;   // 1024*2048 total
    int j = (int)(idx >> 11);
    int k = (int)(idx & 2047);
    float v = (k < GNN_DIM) ? W2l[(size_t)k * GNN_DIM + j]
                            : W2r[(size_t)(k - GNN_DIM) * GNN_DIM + j];
    Wt[idx] = f2bf(v);
}

// ---------------------------------------------------------------------------
// K1: per-edge degree count + layer-1 feature scatter (IN_DIM = 6)
// ---------------------------------------------------------------------------
__global__ void __launch_bounds__(256)
k_edge_l1(const int* __restrict__ src, const int* __restrict__ dst,
          const float* __restrict__ x, float* __restrict__ agg1,
          float* __restrict__ cnt) {
    int e = blockIdx.x * 256 + threadIdx.x;
    if (e >= N_EDGES) return;
    int s = src[e], d = dst[e];
    atomicAdd(&cnt[d], 1.0f);
    #pragma unroll
    for (int k = 0; k < IN_DIM; ++k)
        atomicAdd(&agg1[(size_t)d * IN_DIM + k], x[(size_t)s * IN_DIM + k]);
}

// ---------------------------------------------------------------------------
// K2: layer 1 dense (K=6) + LayerNorm + ReLU -> h1 (bf16).  One block per node.
// ---------------------------------------------------------------------------
__global__ void __launch_bounds__(256)
k_layer1(const float* __restrict__ x, const float* __restrict__ agg1,
         const float* __restrict__ cnt,
         const float* __restrict__ W1l, const float* __restrict__ b1,
         const float* __restrict__ W1r,
         const float* __restrict__ g1, const float* __restrict__ be1,
         unsigned short* __restrict__ h1bf) {
    const int n = blockIdx.x;
    const int t = threadIdx.x;
    float a[IN_DIM], xv[IN_DIM];
    float c = fmaxf(cnt[n], 1.0f);
    #pragma unroll
    for (int k = 0; k < IN_DIM; ++k) {
        xv[k] = x[(size_t)n * IN_DIM + k];
        a[k]  = agg1[(size_t)n * IN_DIM + k] / c;
    }
    float v[4];
    float lsum = 0.f, lsq = 0.f;
    #pragma unroll
    for (int q = 0; q < 4; ++q) {
        int j = q * 256 + t;
        float s = b1[j];
        #pragma unroll
        for (int k = 0; k < IN_DIM; ++k)
            s += a[k] * W1l[(size_t)k * GNN_DIM + j] +
                 xv[k] * W1r[(size_t)k * GNN_DIM + j];
        v[q] = s;
        lsum += s; lsq += s * s;
    }
    __shared__ float s1[256], s2[256];
    s1[t] = lsum; s2[t] = lsq;
    __syncthreads();
    for (int off = 128; off > 0; off >>= 1) {
        if (t < off) { s1[t] += s1[t + off]; s2[t] += s2[t + off]; }
        __syncthreads();
    }
    float mu  = s1[0] * (1.0f / GNN_DIM);
    float var = s2[0] * (1.0f / GNN_DIM) - mu * mu;
    float inv = rsqrtf(var + LN_EPS);
    #pragma unroll
    for (int q = 0; q < 4; ++q) {
        int j = q * 256 + t;
        float h = (v[q] - mu) * inv * g1[j] + be1[j];
        h = fmaxf(h, 0.0f);
        h1bf[(size_t)n * GNN_DIM + j] = f2bf(h);
    }
}

// ---------------------------------------------------------------------------
// K3: layer-2 edge aggregation: agg2[dst] += h1[src].  One block per edge.
// ---------------------------------------------------------------------------
__global__ void __launch_bounds__(256)
k_edge_l2(const int* __restrict__ src, const int* __restrict__ dst,
          const unsigned short* __restrict__ h1bf, float* __restrict__ agg2) {
    const int e = blockIdx.x;
    const int s = src[e], d = dst[e];
    const unsigned short* hp = h1bf + (size_t)s * GNN_DIM;
    float* ap = agg2 + (size_t)d * GNN_DIM;
    #pragma unroll
    for (int q = 0; q < 4; ++q) {
        int j = q * 256 + threadIdx.x;
        atomicAdd(&ap[j], bf2f(hp[j]));
    }
}

// ---------------------------------------------------------------------------
// K4: mean + bf16 convert agg2 -> aggbf
// ---------------------------------------------------------------------------
__global__ void __launch_bounds__(256)
k_aggmean(const float* __restrict__ agg2, const float* __restrict__ cnt,
          unsigned short* __restrict__ aggbf) {
    size_t idx = (size_t)blockIdx.x * 256 + threadIdx.x;  // N_NODES*1024 total
    int n = (int)(idx >> 10);
    float c = fmaxf(cnt[n], 1.0f);
    aggbf[idx] = f2bf(agg2[idx] / c);
}

// ---------------------------------------------------------------------------
// K5: fused layer-2 GEMM with WMMA bf16.
//   out_pre[m][j] = sum_{k<2048} A[m][k] * Wt[j][k] + b2[j]
//   A[m][k] = aggbf[m][k]           (k < 1024)
//           = h1bf [m][k-1024]      (k >= 1024)
// Block: 256 threads = 8 waves; block tile 64(M) x 256(N);
// wave (wm in 0..3, wn in 0..1) computes 16(M) x 128(N) = 8 WMMA tiles.
// ---------------------------------------------------------------------------
__global__ void __launch_bounds__(256)
k_gemm(const unsigned short* __restrict__ Abf0,   // aggbf [M_PAD][1024]
       const unsigned short* __restrict__ Abf1,   // h1bf  [M_PAD][1024]
       const unsigned short* __restrict__ Wt,     // [1024][2048]
       const float* __restrict__ b2,
       float* __restrict__ out_pre)               // [M_PAD][1024]
{
    const int lane = threadIdx.x & 31;
    const int wave = threadIdx.x >> 5;
    const int wm = wave >> 1, wn = wave & 1;
    const int half = lane >> 4, l15 = lane & 15;
    const int mBase = blockIdx.y * 64 + wm * 16;
    const int nBase = blockIdx.x * 256 + wn * 128;
    const int rowA = mBase + l15;

    f32x8 acc[8];
    #pragma unroll
    for (int s = 0; s < 8; ++s)
        #pragma unroll
        for (int i = 0; i < 8; ++i) acc[s][i] = 0.0f;

    const int o0 = half * 8;        // byte-aligned 16B fragment chunk offsets
    const int o1 = 16 + half * 8;

    for (int kk = 0; kk < CAT_DIM; kk += 32) {
        const unsigned short* Aptr =
            (kk < GNN_DIM) ? (Abf0 + (size_t)rowA * GNN_DIM + kk)
                           : (Abf1 + (size_t)rowA * GNN_DIM + (kk - GNN_DIM));
        bf16x16 afrag = load_frag(Aptr + o0, Aptr + o1);
        #pragma unroll
        for (int s = 0; s < 8; ++s) {
            const unsigned short* Bptr =
                Wt + (size_t)(nBase + s * 16 + l15) * CAT_DIM + kk;
            bf16x16 bfrag = load_frag(Bptr + o0, Bptr + o1);
            acc[s] = __builtin_amdgcn_wmma_f32_16x16x32_bf16(
                false, afrag, false, bfrag, (short)0, acc[s], false, false);
        }
    }

    #pragma unroll
    for (int s = 0; s < 8; ++s) {
        int col = nBase + s * 16 + l15;
        float bias = b2[col];
        #pragma unroll
        for (int i = 0; i < 8; ++i) {
            int row = mBase + half * 8 + i;     // C layout: lanes16-31 -> M=i+8
            out_pre[(size_t)row * GNN_DIM + col] = acc[s][i] + bias;
        }
    }
}

// ---------------------------------------------------------------------------
// K6: LayerNorm + ReLU of layer-2 output, fused with graph mean-pool (sums)
// ---------------------------------------------------------------------------
__global__ void __launch_bounds__(256)
k_ln2_pool(const float* __restrict__ out_pre,
           const float* __restrict__ g2, const float* __restrict__ be2,
           const int* __restrict__ batch,
           float* __restrict__ gsum, float* __restrict__ gcnt) {
    const int n = blockIdx.x;
    const int t = threadIdx.x;
    const float* row = out_pre + (size_t)n * GNN_DIM;
    float v[4];
    float lsum = 0.f, lsq = 0.f;
    #pragma unroll
    for (int q = 0; q < 4; ++q) {
        v[q] = row[q * 256 + t];
        lsum += v[q]; lsq += v[q] * v[q];
    }
    __shared__ float s1[256], s2[256];
    s1[t] = lsum; s2[t] = lsq;
    __syncthreads();
    for (int off = 128; off > 0; off >>= 1) {
        if (t < off) { s1[t] += s1[t + off]; s2[t] += s2[t + off]; }
        __syncthreads();
    }
    float mu  = s1[0] * (1.0f / GNN_DIM);
    float var = s2[0] * (1.0f / GNN_DIM) - mu * mu;
    float inv = rsqrtf(var + LN_EPS);
    int b = batch[n];
    float* gs = gsum + (size_t)b * CAT_DIM;
    #pragma unroll
    for (int q = 0; q < 4; ++q) {
        int j = q * 256 + t;
        float h = fmaxf((v[q] - mu) * inv * g2[j] + be2[j], 0.0f);
        atomicAdd(&gs[j], h);
    }
    if (t == 0) atomicAdd(&gcnt[b], 1.0f);
}

// ---------------------------------------------------------------------------
// K7: multi-scale Fourier features, pooled directly into graph sums
// ---------------------------------------------------------------------------
__global__ void __launch_bounds__(128)
k_ffm_pool(const float* __restrict__ coords, const float* __restrict__ Bm,
           const int* __restrict__ batch, float* __restrict__ gsum) {
    const int n = blockIdx.x;
    const int j = threadIdx.x;              // 0..127
    float c0 = coords[(size_t)n * 3 + 0];
    float c1 = coords[(size_t)n * 3 + 1];
    float c2 = coords[(size_t)n * 3 + 2];
    float d = c0 * Bm[j] + c1 * Bm[FFM_DIM + j] + c2 * Bm[2 * FFM_DIM + j];
    int b = batch[n];
    float* gs = gsum + (size_t)b * CAT_DIM + GNN_DIM;
    const float scales[4] = {1.0f, 5.0f, 10.0f, 30.0f};
    #pragma unroll
    for (int s = 0; s < 4; ++s) {
        float p = d * scales[s];
        float sn, cs;
        __sincosf(p, &sn, &cs);
        atomicAdd(&gs[s * 256 + j], sn);
        atomicAdd(&gs[s * 256 + 128 + j], cs);
    }
}

// ---------------------------------------------------------------------------
// K8: final FC (2048 -> 256) + LayerNorm.  One block per graph.
// ---------------------------------------------------------------------------
__global__ void __launch_bounds__(256)
k_final(const float* __restrict__ gsum, const float* __restrict__ gcnt,
        const float* __restrict__ Wfc, const float* __restrict__ bfc,
        const float* __restrict__ gfc, const float* __restrict__ befc,
        float* __restrict__ out) {
    const int g = blockIdx.x;
    const int t = threadIdx.x;
    __shared__ float hg[CAT_DIM];
    float c = fmaxf(gcnt[g], 1.0f);
    #pragma unroll
    for (int q = 0; q < 8; ++q) {
        int j = q * 256 + t;
        hg[j] = gsum[(size_t)g * CAT_DIM + j] / c;
    }
    __syncthreads();
    float y = bfc[t];
    for (int j = 0; j < CAT_DIM; ++j)
        y += hg[j] * Wfc[(size_t)j * OUT_DIM + t];
    __shared__ float s1[256], s2[256];
    s1[t] = y; s2[t] = y * y;
    __syncthreads();
    for (int off = 128; off > 0; off >>= 1) {
        if (t < off) { s1[t] += s1[t + off]; s2[t] += s2[t + off]; }
        __syncthreads();
    }
    float mu  = s1[0] * (1.0f / OUT_DIM);
    float var = s2[0] * (1.0f / OUT_DIM) - mu * mu;
    float inv = rsqrtf(var + LN_EPS);
    out[(size_t)g * OUT_DIM + t] = (y - mu) * inv * gfc[t] + befc[t];
}

// ---------------------------------------------------------------------------
// host launcher
// ---------------------------------------------------------------------------
static inline size_t align256(size_t x) { return (x + 255) & ~(size_t)255; }

extern "C" void kernel_launch(void* const* d_in, const int* in_sizes, int n_in,
                              void* d_out, int out_size, void* d_ws, size_t ws_size,
                              hipStream_t stream) {
    const float* x      = (const float*)d_in[0];
    const int*   eidx   = (const int*)  d_in[1];
    const float* coords = (const float*)d_in[2];
    const int*   batch  = (const int*)  d_in[3];
    const float* Bm     = (const float*)d_in[4];
    const float* W1l    = (const float*)d_in[5];
    const float* b1     = (const float*)d_in[6];
    const float* W1r    = (const float*)d_in[7];
    const float* g1     = (const float*)d_in[8];
    const float* be1    = (const float*)d_in[9];
    const float* W2l    = (const float*)d_in[10];
    const float* b2     = (const float*)d_in[11];
    const float* W2r    = (const float*)d_in[12];
    const float* g2     = (const float*)d_in[13];
    const float* be2    = (const float*)d_in[14];
    const float* Wfc    = (const float*)d_in[15];
    const float* bfc    = (const float*)d_in[16];
    const float* gfc    = (const float*)d_in[17];
    const float* befc   = (const float*)d_in[18];
    float* out = (float*)d_out;

    const int* src = eidx;
    const int* dst = eidx + N_EDGES;

    // workspace layout
    char* ws = (char*)d_ws;
    size_t o = 0;
    float*          cnt    = (float*)(ws + o);          o += align256((size_t)N_NODES * 4);
    float*          agg1   = (float*)(ws + o);          o += align256((size_t)N_NODES * IN_DIM * 4);
    unsigned short* h1bf   = (unsigned short*)(ws + o); o += align256((size_t)M_PAD * GNN_DIM * 2);
    unsigned short* aggbf  = (unsigned short*)(ws + o); o += align256((size_t)M_PAD * GNN_DIM * 2);
    float*          agg2   = (float*)(ws + o);          o += align256((size_t)M_PAD * GNN_DIM * 4);
    float*          outpre = agg2;   // aliased: agg2 dead once aggbf built
    unsigned short* Wt     = (unsigned short*)(ws + o); o += align256((size_t)GNN_DIM * CAT_DIM * 2);
    float*          gsum   = (float*)(ws + o);          o += align256((size_t)N_GRAPHS * CAT_DIM * 4);
    float*          gcnt   = (float*)(ws + o);          o += align256((size_t)N_GRAPHS * 4);
    (void)ws_size; (void)n_in; (void)in_sizes; (void)out_size;

    // zero the accumulation regions (every call; graph-capture safe)
    hipMemsetAsync(cnt,  0, (size_t)N_NODES * 4, stream);
    hipMemsetAsync(agg1, 0, (size_t)N_NODES * IN_DIM * 4, stream);
    hipMemsetAsync(agg2, 0, (size_t)N_NODES * GNN_DIM * 4, stream);
    hipMemsetAsync(h1bf  + (size_t)N_NODES * GNN_DIM, 0,
                   (size_t)(M_PAD - N_NODES) * GNN_DIM * 2, stream);
    hipMemsetAsync(aggbf + (size_t)N_NODES * GNN_DIM, 0,
                   (size_t)(M_PAD - N_NODES) * GNN_DIM * 2, stream);
    hipMemsetAsync(gsum, 0, (size_t)N_GRAPHS * CAT_DIM * 4, stream);
    hipMemsetAsync(gcnt, 0, (size_t)N_GRAPHS * 4, stream);

    // 0) weight transpose + bf16
    k_prepW<<<(GNN_DIM * CAT_DIM) / 256, 256, 0, stream>>>(W2l, W2r, Wt);
    // 1) degree + layer-1 scatter
    k_edge_l1<<<(N_EDGES + 255) / 256, 256, 0, stream>>>(src, dst, x, agg1, cnt);
    // 2) layer-1 dense + LN + ReLU -> h1 bf16
    k_layer1<<<N_NODES, 256, 0, stream>>>(x, agg1, cnt, W1l, b1, W1r, g1, be1, h1bf);
    // 3) layer-2 aggregation (gather/scatter)
    k_edge_l2<<<N_EDGES, 256, 0, stream>>>(src, dst, h1bf, agg2);
    // 4) mean + bf16
    k_aggmean<<<(N_NODES * GNN_DIM) / 256, 256, 0, stream>>>(agg2, cnt, aggbf);
    // 5) fused layer-2 GEMM via WMMA
    {
        dim3 grid(GNN_DIM / 256, M_PAD / 64);
        k_gemm<<<grid, 256, 0, stream>>>(aggbf, h1bf, Wt, b2, outpre);
    }
    // 6) LN + ReLU + graph pooling
    k_ln2_pool<<<N_NODES, 256, 0, stream>>>(outpre, g2, be2, batch, gsum, gcnt);
    // 7) Fourier features pooled
    k_ffm_pool<<<N_NODES, FFM_DIM, 0, stream>>>(coords, Bm, batch, gsum);
    // 8) final FC + LN
    k_final<<<N_GRAPHS, 256, 0, stream>>>(gsum, gcnt, Wfc, bfc, gfc, befc, out);
}
